// QLSTMEnhanced_65481071410302
// MI455X (gfx1250) — matrix operand order repaired and checked by reading, not compiled, not run on hardware
//
#include <hip/hip_runtime.h>
#include <math.h>

// ---------------------------------------------------------------------------
// QLSTM on MI455X (gfx1250, wave32, WMMA).
//
// Per sample, per gate, per timestep the quantum circuit is:
//   phi = V * (D * r),  r = kron_w [cos(z_w/2), sin(z_w/2)],  D = diag((-i)^popcount)
// with V a fixed REAL 256x256 matrix determined only by the gate params
// (two RY layers + CNOT chains).  Folding D's phase into r gives two real
// GEMMs sharing the same B operand:
//   phi_re = V * r_re,  phi_im = V * r_im,   probs = phi_re^2 + phi_im^2
// and the PauliZ expvals are probs @ S with S a fixed +-1 (256x8) matrix.
// All three become v_wmma_f32_16x16x32_f16 with M = 16 samples per wave.
// Samples are independent across time, so one wave owns 16 samples for all
// 256 timesteps (512 blocks x 32 threads, no global sync).
// ---------------------------------------------------------------------------

typedef __attribute__((ext_vector_type(16))) _Float16 v16h;
typedef __attribute__((ext_vector_type(8)))  _Float16 v8h;
typedef __attribute__((ext_vector_type(8)))  float    v8f;

static __device__ inline v8f wmma_f16(v16h a, v16h b, v8f c) {
  return __builtin_amdgcn_wmma_f32_16x16x32_f16(false, a, false, b, (short)0, c,
                                                false, false);
}

// ---------------------------------------------------------------------------
// Precompute V^T tiles in WMMA B-operand layout.
// Block (g,k) simulates circuit column k of gate g in LDS (256 floats).
// B tile (kt,jt): lane l holds N = l%16, K = (l<16 ? 0..7,16..23 : 8..15,24..31) of the
// 32x16 tile; element e = (K&7) + 8*(K>>4).  Vt[g][jt][kt][lane][e] = V[n][k].
// ---------------------------------------------------------------------------
__global__ __launch_bounds__(256)
void qlstm_build_vt(const float* __restrict__ p0, const float* __restrict__ p1,
                    const float* __restrict__ p2, const float* __restrict__ p3,
                    _Float16* __restrict__ Vt)
{
  const int g = blockIdx.x >> 8;
  const int k = blockIdx.x & 255;
  const float* p = (g == 0) ? p0 : (g == 1) ? p1 : (g == 2) ? p2 : p3;
  __shared__ float bufA[256], bufB[256];
  const int j = threadIdx.x;
  bufA[j] = (j == k) ? 1.0f : 0.0f;
  __syncthreads();
  float* cur = bufA;
  float* nxt = bufB;
  for (int layer = 0; layer < 2; ++layer) {
    for (int w = 0; w < 8; ++w) {              // RY(params[layer*8+w]) on wire w
      float th = 0.5f * p[layer * 8 + w];
      float cr = __cosf(th), sr = __sinf(th);
      int msk = 1 << (7 - w);                  // wire w <-> bit (7-w)
      float v0 = cur[j & ~msk];
      float v1 = cur[j |  msk];
      float nv = ((j & msk) == 0) ? (cr * v0 - sr * v1) : (sr * v0 + cr * v1);
      nxt[j] = nv;
      __syncthreads();
      float* tp = cur; cur = nxt; nxt = tp;
    }
    // CNOT chain ctrl=0..6: new[j] = old[m0(m1(...m6(j)))]
    int src = j;
    for (int c = 6; c >= 0; --c) src ^= ((src >> (7 - c)) & 1) << (6 - c);
    float nv = cur[src];
    nxt[j] = nv;
    __syncthreads();
    float* tp = cur; cur = nxt; nxt = tp;
  }
  // cur[j] == V[j][k] -> scatter into B-operand layout
  int jt = j >> 4, nn = j & 15;
  int kt = k >> 5, kk = k & 31;
  int lane = ((kk >> 3) & 1) ? (16 + nn) : nn;
  int e = (kk & 7) + ((kk >> 4) << 3);
  size_t idx = ((((size_t)g * 16 + jt) * 8 + kt) * 32 + lane) * 16 + e;
  Vt[idx] = (_Float16)cur[j];
}

// Sign matrix S[k][w] = 1 - 2*bit_{7-w}(k) (w<8, else 0), in B-operand layout.
__global__ __launch_bounds__(256)
void qlstm_build_sgn(_Float16* __restrict__ S)
{
  int tid = blockIdx.x * 256 + threadIdx.x;
  if (tid >= 8 * 32 * 16) return;
  int e = tid & 15;
  int lane = (tid >> 4) & 31;
  int kt = tid >> 9;
  int nn = lane & 15;
  int kk = ((lane < 16) ? 0 : 8) + (e & 7) + ((e >> 3) << 4);
  int k = kt * 32 + kk;
  float v = 0.0f;
  if (nn < 8) v = 1.0f - 2.0f * (float)((k >> (7 - nn)) & 1);
  S[tid] = (_Float16)v;
}

// ---------------------------------------------------------------------------
// Main persistent kernel: 1 wave = 16 samples for all 256 timesteps.
// ---------------------------------------------------------------------------
__global__ __launch_bounds__(32)
void qlstm_main(const float* __restrict__ x,
                const float* __restrict__ Wf, const float* __restrict__ bf,
                const float* __restrict__ Wi, const float* __restrict__ bi,
                const float* __restrict__ Wu, const float* __restrict__ bu,
                const float* __restrict__ Wo, const float* __restrict__ bo,
                const _Float16* __restrict__ Vt, const _Float16* __restrict__ Sg,
                float* __restrict__ out)
{
  constexpr int T = 256, NB = 8192, DI = 16, H = 8;
  __shared__ __align__(16) float Wl[4][8][24];
  __shared__ float bl[4][8];
  __shared__ __align__(16) float combx[16][16];
  __shared__ float hxs[16][8];
  __shared__ float cxs[16][8];
  __shared__ float zbuf[16][8];
  __shared__ float ev[4][16][8];
  __shared__ __align__(16) _Float16 pbuf[16][264];   // padded rows (bank spread)

  const int l  = threadIdx.x;
  const int sb = blockIdx.x * 16;
  const int m  = l & 15;          // sample (A/D row group) for this lane
  const int hb = l >> 4;          // bit3 of k handled by this lane half
  const bool hbb = (hb != 0);

  { const float* Ws[4] = {Wf, Wi, Wu, Wo};
    const float* bs[4] = {bf, bi, bu, bo};
    for (int g = 0; g < 4; ++g) {
      for (int idx = l; idx < 192; idx += 32) (&Wl[g][0][0])[idx] = Ws[g][idx];
      if (l < 8) bl[g][l] = bs[g][l];
    }
  }
  for (int idx = l; idx < 128; idx += 32) {
    (&hxs[0][0])[idx] = 0.0f;
    (&cxs[0][0])[idx] = 0.0f;
  }
  __syncthreads();

  #pragma unroll 1
  for (int t = 0; t < T; ++t) {
    { // stage x_t for our 16 samples (256 contiguous floats)
      const float4* xp = (const float4*)(x + ((size_t)t * NB + sb) * DI);
      float4* cb = (float4*)(&combx[0][0]);
      cb[l * 2]     = xp[l * 2];
      cb[l * 2 + 1] = xp[l * 2 + 1];
    }
    __syncthreads();

    #pragma unroll 1
    for (int g = 0; g < 4; ++g) {
      // ---- z = 0.5*(comb @ W^T + b): 4 dot products per lane -------------
      { int s = l >> 1, ob = (l & 1) * 4;
        #pragma unroll
        for (int oo = 0; oo < 4; ++oo) {
          int o = ob + oo;
          float acc = bl[g][o];
          #pragma unroll
          for (int d = 0; d < 16; ++d) acc += combx[s][d] * Wl[g][o][d];
          #pragma unroll
          for (int h = 0; h < 8; ++h)  acc += hxs[s][h] * Wl[g][o][16 + h];
          zbuf[s][o] = 0.5f * acc;
        }
      }
      __syncthreads();

      // ---- build A operands r_re / r_im for my sample, my k half ---------
      float cw[8], sw[8];
      #pragma unroll
      for (int w = 0; w < 8; ++w) {
        float a = zbuf[m][w];
        cw[w] = __cosf(a);
        sw[w] = __sinf(a);
      }
      float prod3[8], arr8[8];
      #pragma unroll
      for (int q = 0; q < 8; ++q) {
        prod3[q] = ((q & 4) ? sw[0] : cw[0]) * ((q & 2) ? sw[1] : cw[1]) *
                   ((q & 1) ? sw[2] : cw[2]);                 // k bits 7..5
        arr8[q]  = ((q & 4) ? sw[5] : cw[5]) * ((q & 2) ? sw[6] : cw[6]) *
                   ((q & 1) ? sw[7] : cw[7]);                 // k bits 2..0
      }
      float f3 = hbb ? sw[4] : cw[4];                          // k bit 3 (fixed)

      v16h are[8], aim[8];
      #pragma unroll
      for (int kt = 0; kt < 8; ++kt) {
        float bk = prod3[kt] * f3;
        #pragma unroll
        for (int b4 = 0; b4 < 2; ++b4) {
          float base = bk * (b4 ? sw[3] : cw[3]);             // k bit 4
          #pragma unroll
          for (int lo = 0; lo < 8; ++lo) {
            float r = base * arr8[lo];
            int pcf = ((kt >> 2) & 1) + ((kt >> 1) & 1) + (kt & 1) + b4 +
                      ((lo >> 2) & 1) + ((lo >> 1) & 1) + (lo & 1);
            int m0 = pcf & 3;                                 // pc w/o bit3
            float reV, imV;                                   // phase (-i)^pc
            if      (m0 == 0) { reV = hbb ? 0.f : r;   imV = hbb ? -r : 0.f; }
            else if (m0 == 1) { reV = hbb ? -r : 0.f;  imV = hbb ? 0.f : -r; }
            else if (m0 == 2) { reV = hbb ? 0.f : -r;  imV = hbb ? r : 0.f; }
            else              { reV = hbb ? r : 0.f;   imV = hbb ? 0.f : r; }
            are[kt][b4 * 8 + lo] = (_Float16)reV;
            aim[kt][b4 * 8 + lo] = (_Float16)imV;
          }
        }
      }

      // ---- phi_re/phi_im GEMMs (shared B tiles), probs -> pbuf -----------
      const _Float16* vg = Vt + (size_t)g * (16 * 8 * 32 * 16);
      #pragma unroll 1
      for (int jt = 0; jt < 16; ++jt) {
        v8f dre = {};
        v8f dim2 = {};
        const _Float16* vb = vg + ((size_t)jt * 8 * 32 + l) * 16;
        #pragma unroll
        for (int kt = 0; kt < 8; ++kt) {
          v16h bt = *(const v16h*)(vb + (size_t)kt * 512);
          dre  = wmma_f16(are[kt], bt, dre);
          dim2 = wmma_f16(aim[kt], bt, dim2);
        }
        int kcol = jt * 16 + m;   // D layout: lane%16 = N, vgpr r = M row
        #pragma unroll
        for (int r = 0; r < 8; ++r) {
          float pv = dre[r] * dre[r] + dim2[r] * dim2[r];
          pbuf[hb * 8 + r][kcol] = (_Float16)pv;
        }
      }
      __syncthreads();

      // ---- measurement GEMM: expvals = probs @ S -------------------------
      { v8f dm = {};
        const _Float16* pr = &pbuf[m][0];
        #pragma unroll
        for (int kt = 0; kt < 8; ++kt) {
          int k0 = kt * 32 + hb * 8;
          v8h plo = *(const v8h*)(pr + k0);
          v8h phi = *(const v8h*)(pr + k0 + 16);
          v16h pa = __builtin_shufflevector(plo, phi, 0, 1, 2, 3, 4, 5, 6, 7,
                                            8, 9, 10, 11, 12, 13, 14, 15);
          v16h st = *(const v16h*)(Sg + ((size_t)kt * 32 + l) * 16);
          dm = wmma_f16(pa, st, dm);
        }
        if (m < 8) {              // lane%16 = w column; vgpr r = sample row
          #pragma unroll
          for (int r = 0; r < 8; ++r) ev[g][hb * 8 + r][m] = dm[r];
        }
      }
      __syncthreads();
    } // gate loop

    // ---- LSTM cell update + output store ---------------------------------
    { float hv[4];
      #pragma unroll
      for (int q = 0; q < 4; ++q) {
        int flat = l * 4 + q, s = flat >> 3, h = flat & 7;
        float fz = 1.0f / (1.0f + __expf(-ev[0][s][h]));
        float iz = 1.0f / (1.0f + __expf(-ev[1][s][h]));
        float gz = tanhf(ev[2][s][h]);
        float oz = 1.0f / (1.0f + __expf(-ev[3][s][h]));
        float c  = fz * cxs[s][h] + iz * gz;
        float hh = oz * tanhf(c);
        cxs[s][h] = c;
        hxs[s][h] = hh;
        hv[q] = hh;
      }
      float4* op = (float4*)(out + ((size_t)t * NB + sb) * H);
      op[l] = make_float4(hv[0], hv[1], hv[2], hv[3]);
    }
    __syncthreads();
  } // t loop

  { size_t base = (size_t)T * NB * H;
    #pragma unroll
    for (int q = 0; q < 4; ++q) {
      int flat = l * 4 + q, s = flat >> 3, h = flat & 7;
      out[base + (size_t)(sb + s) * H + h] = hxs[s][h];
      out[base + (size_t)NB * H + (size_t)(sb + s) * H + h] = cxs[s][h];
    }
  }
}

extern "C" void kernel_launch(void* const* d_in, const int* in_sizes, int n_in,
                              void* d_out, int out_size, void* d_ws, size_t ws_size,
                              hipStream_t stream)
{
  const float* x  = (const float*)d_in[0];
  const float* Wf = (const float*)d_in[1];
  const float* bf = (const float*)d_in[2];
  const float* Wi = (const float*)d_in[3];
  const float* bi = (const float*)d_in[4];
  const float* Wu = (const float*)d_in[5];
  const float* bu = (const float*)d_in[6];
  const float* Wo = (const float*)d_in[7];
  const float* bo = (const float*)d_in[8];
  const float* pf = (const float*)d_in[9];
  const float* pi = (const float*)d_in[10];
  const float* pu = (const float*)d_in[11];
  const float* po = (const float*)d_in[12];
  float* out = (float*)d_out;

  _Float16* Vt = (_Float16*)d_ws;                                   // 512 KB
  _Float16* Sg = (_Float16*)((char*)d_ws + (size_t)4 * 16 * 8 * 32 * 16 * 2);

  qlstm_build_vt<<<1024, 256, 0, stream>>>(pf, pi, pu, po, Vt);
  qlstm_build_sgn<<<16, 256, 0, stream>>>(Sg);
  qlstm_main<<<512, 32, 0, stream>>>(x, Wf, bf, Wi, bi, Wu, bu, Wo, bo,
                                     Vt, Sg, out);
  (void)in_sizes; (void)n_in; (void)out_size; (void)ws_size;
}
